// WindowedValuesCoordinatesAttention_90898687853173
// MI455X (gfx1250) — compile-verified, matrix operand order checked by reading, not run on hardware
//
#include <hip/hip_runtime.h>
#include <hip/hip_bf16.h>

// ---------- problem constants ----------
#define BB    4
#define SS    16
#define TT    256
#define DV    256
#define DC    32
#define INNERD 256
#define HH    8
#define DH    32
#define WSZ   4
#define NWIN  16      // BB * (SS/WSZ)
#define LTOK  1024    // WSZ * TT

typedef __attribute__((ext_vector_type(16))) __bf16 v16bf;
typedef __attribute__((ext_vector_type(8)))  __bf16 bf16x8;
typedef __attribute__((ext_vector_type(8)))  float  v8f;

// ---------- workspace layout (bytes) ----------
// idx:    64 ints                 @ 0
// mw:     NWIN*LTOK bytes         @ 1024
// WqkvT:  768*256 bf16 (384KB)    @ 0x8000
// WqkcT:  512*32  bf16 (32KB)
// WoutT:  256*256 bf16 (128KB)
// Q:  NWIN*H*1024*64 bf16 (16MB)  (clamped, *scale, [qp|qc] concat on d)
// K:  NWIN*H*1024*64 bf16 (16MB)  (clamped, [kp|kc])
// Vt: NWIN*H*32*1024 bf16 (8MB)   (transposed: [d][key])
// O:  NWIN*1024*256 bf16 (8MB)    (attention output, heads concat)
static const size_t OFF_IDX   = 0;
static const size_t OFF_MW    = 1024;
static const size_t OFF_WQKVT = 0x8000;
static const size_t OFF_WQKCT = OFF_WQKVT + (size_t)768*256*2;
static const size_t OFF_WOUTT = OFF_WQKCT + (size_t)512*32*2;
static const size_t OFF_Q     = (OFF_WOUTT + (size_t)256*256*2 + 255) & ~(size_t)255;
static const size_t OFF_K     = OFF_Q  + (size_t)NWIN*HH*LTOK*64*2;
static const size_t OFF_VT    = OFF_K  + (size_t)NWIN*HH*LTOK*64*2;
static const size_t OFF_O     = OFF_VT + (size_t)NWIN*HH*DH*LTOK*2;

// ---------- WMMA helpers ----------
__device__ __forceinline__ v16bf mk16(bf16x8 lo, bf16x8 hi) {
  v16bf r;
#pragma unroll
  for (int i = 0; i < 8; ++i) { r[i] = lo[i]; r[i + 8] = hi[i]; }
  return r;
}

// Read one WMMA 16x32 bf16 operand (A or B) from a row-major tile.
// Lane l: row = rowBase + (l&15); two 16B runs at elem offsets kOff+(l>>4)*8
// and kOff+16+(l>>4)*8 — matches the CDNA5 16-bit A/B VGPR layout.
__device__ __forceinline__ v16bf ld_op(const __bf16* base, int rowBase,
                                       int rowStride, int kOff, int lane) {
  int m = lane & 15, h = lane >> 4;
  const __bf16* p = base + (size_t)(rowBase + m) * rowStride + kOff + h * 8;
  bf16x8 lo = *(const bf16x8*)p;
  bf16x8 hi = *(const bf16x8*)(p + 16);
  return mk16(lo, hi);
}

__device__ __forceinline__ v8f wmma_bf16(v16bf a, v16bf b, v8f c) {
  return __builtin_amdgcn_wmma_f32_16x16x32_bf16(false, a, false, b,
                                                 (short)0, c, false, false);
}

__device__ __forceinline__ float clamp5(float x) {
  return fminf(fmaxf(x, -5.0f), 5.0f);
}

// ---------- kernel 0: argsort(dt) ranks + gathered/rolled window mask ----------
__global__ void k_prep(const float* __restrict__ dt,
                       const unsigned char* __restrict__ mask,
                       int* __restrict__ idx, unsigned char* __restrict__ mw) {
  int tid = threadIdx.x;
  if (tid < BB * SS) {
    int b = tid / SS, s = tid % SS;
    float v = dt[b * SS + s];
    int r = 0;
    for (int s2 = 0; s2 < SS; ++s2) {
      float v2 = dt[b * SS + s2];
      r += (v2 < v) || (v2 == v && s2 < s);   // stable argsort rank
    }
    idx[b * SS + r] = s;
  }
  __syncthreads();
  for (int i = tid; i < NWIN * LTOK; i += blockDim.x) {
    int win = i / LTOK, pos = i % LTOK;
    int b = win / WSZ, w = win % WSZ;
    int p = pos / TT, t2 = pos % TT;
    int q = w * WSZ + p;
    int s = idx[b * SS + ((q + SS - 2) & (SS - 1))];   // roll by +ws/2
    mw[i] = mask[(b * SS + s) * TT + t2];
  }
}

// ---------- kernel 1: fp32 -> bf16 transposed weight copies ----------
__global__ void k_weights(const float* __restrict__ Wqkv,
                          const float* __restrict__ Wqkc,
                          const float* __restrict__ Wout,
                          __bf16* __restrict__ wqkvT,
                          __bf16* __restrict__ wqkcT,
                          __bf16* __restrict__ woutT) {
  int i = blockIdx.x * blockDim.x + threadIdx.x;
  const int n1 = 768 * 256, n2 = 512 * 32, n3 = 256 * 256;
  if (i < n1) {
    int c = i / 256, k = i % 256;
    wqkvT[i] = (__bf16)Wqkv[k * 768 + c];
  } else if (i < n1 + n2) {
    int j = i - n1; int c = j / 32, k = j % 32;
    wqkcT[j] = (__bf16)Wqkc[k * 512 + c];
  } else if (i < n1 + n2 + n3) {
    int j = i - n1 - n2; int c = j / 256, k = j % 256;
    woutT[j] = (__bf16)Wout[k * 256 + c];
  }
}

// ---------- kernel 2: gathered projections (WMMA GEMM) ----------
// block = (window, 64-row group); 128 threads = 4 waves
__global__ __launch_bounds__(128) void k_proj(
    const float* __restrict__ values, const float* __restrict__ coords,
    const int* __restrict__ idx,
    const __bf16* __restrict__ wqkvT, const __bf16* __restrict__ wqkcT,
    __bf16* __restrict__ Qb, __bf16* __restrict__ Kb, __bf16* __restrict__ Vt) {
  __shared__ __align__(16) __bf16 Xv[64 * 256];   // 32KB
  __shared__ __align__(16) __bf16 Xc[64 * 32];    // 4KB
  __shared__ __align__(16) __bf16 Wt[32 * 256];   // 16KB

  int blk = blockIdx.x;
  int win = blk / 16, mg = blk % 16;
  int b = win / WSZ, w = win % WSZ;
  int rowBase = mg * 64;
  int p = rowBase / TT;                 // window slot 0..3 (group fits one slot)
  int ttBase = rowBase % TT;
  int srcS = idx[b * SS + ((w * WSZ + p + SS - 2) & (SS - 1))];
  int tid = threadIdx.x, lane = tid & 31, wv = tid >> 5;
  const float scale = 0.17677669529663687f;       // dh^-0.5

  // stage gathered values (fp32 -> bf16) into LDS, row-major [64][256]
  const float* vsrc = values + (((size_t)(b * SS + srcS)) * TT + ttBase) * DV;
  for (int e = tid; e < 64 * (DV / 4); e += blockDim.x) {
    int row = e / (DV / 4), k4 = (e % (DV / 4)) * 4;
    float4 f = *(const float4*)(vsrc + (size_t)row * DV + k4);
    __bf16* d = &Xv[row * DV + k4];
    d[0] = (__bf16)f.x; d[1] = (__bf16)f.y; d[2] = (__bf16)f.z; d[3] = (__bf16)f.w;
  }
  const float* csrc = coords + (((size_t)(b * SS + srcS)) * TT + ttBase) * DC;
  for (int e = tid; e < 64 * (DC / 4); e += blockDim.x) {
    int row = e / (DC / 4), k4 = (e % (DC / 4)) * 4;
    float4 f = *(const float4*)(csrc + (size_t)row * DC + k4);
    __bf16* d = &Xc[row * DC + k4];
    d[0] = (__bf16)f.x; d[1] = (__bf16)f.y; d[2] = (__bf16)f.z; d[3] = (__bf16)f.w;
  }
  __syncthreads();

  // ---- X(64x256) @ Wqkv(256x768): 24 N-groups of 32 ----
  for (int ng = 0; ng < 24; ++ng) {
    for (int e = tid; e < 32 * 256 / 8; e += blockDim.x)
      ((uint4*)Wt)[e] = ((const uint4*)(wqkvT + (size_t)ng * 32 * 256))[e];
    __syncthreads();
#pragma unroll
    for (int j = 0; j < 2; ++j) {
      int t8 = wv * 2 + j, mt = t8 >> 1, nt = t8 & 1;
      v8f acc = {};
#pragma unroll
      for (int kc = 0; kc < 8; ++kc)
        acc = wmma_bf16(ld_op(Xv, mt * 16, 256, kc * 32, lane),
                        ld_op(Wt, nt * 16, 256, kc * 32, lane), acc);
      int n = lane & 15, h = lane >> 4;
      int c = ng * 32 + nt * 16 + n;
#pragma unroll
      for (int r = 0; r < 8; ++r) {
        int row = rowBase + mt * 16 + r + 8 * h;
        float x = acc[r];
        if (c < 256) {            // q_p -> clamp * scale
          int hh = c >> 5, d = c & 31;
          Qb[(((size_t)win * HH + hh) * LTOK + row) * 64 + d] =
              (__bf16)(clamp5(x) * scale);
        } else if (c < 512) {     // k_p -> clamp
          int c2 = c - 256, hh = c2 >> 5, d = c2 & 31;
          Kb[(((size_t)win * HH + hh) * LTOK + row) * 64 + d] = (__bf16)clamp5(x);
        } else {                  // v -> transposed [d][key]
          int c2 = c - 512, hh = c2 >> 5, d = c2 & 31;
          Vt[(((size_t)win * HH + hh) * DH + d) * LTOK + row] = (__bf16)x;
        }
      }
    }
    __syncthreads();
  }

  // ---- Xc(64x32) @ Wqkc(32x512): 16 N-groups of 32, single K step ----
  for (int ng = 0; ng < 16; ++ng) {
    for (int e = tid; e < 32 * 32 / 8; e += blockDim.x)
      ((uint4*)Wt)[e] = ((const uint4*)(wqkcT + (size_t)ng * 32 * 32))[e];
    __syncthreads();
#pragma unroll
    for (int j = 0; j < 2; ++j) {
      int t8 = wv * 2 + j, mt = t8 >> 1, nt = t8 & 1;
      v8f acc = {};
      acc = wmma_bf16(ld_op(Xc, mt * 16, 32, 0, lane),
                      ld_op(Wt, nt * 16, 32, 0, lane), acc);
      int n = lane & 15, h = lane >> 4;
      int c = ng * 32 + nt * 16 + n;
#pragma unroll
      for (int r = 0; r < 8; ++r) {
        int row = rowBase + mt * 16 + r + 8 * h;
        float y = clamp5(acc[r]);
        if (c < 256) {            // q_c (concat at d+32)
          int hh = c >> 5, d = c & 31;
          Qb[(((size_t)win * HH + hh) * LTOK + row) * 64 + 32 + d] =
              (__bf16)(y * scale);
        } else {                  // k_c
          int c2 = c - 256, hh = c2 >> 5, d = c2 & 31;
          Kb[(((size_t)win * HH + hh) * LTOK + row) * 64 + 32 + d] = (__bf16)y;
        }
      }
    }
    __syncthreads();
  }
}

// ---------- kernel 3: flash attention per (window, head) ----------
// block = (win, head, 64-q group); 128 threads = 4 waves, one 16-q tile each
__global__ __launch_bounds__(128) void k_attn(
    const __bf16* __restrict__ Qb, const __bf16* __restrict__ Kb,
    const __bf16* __restrict__ Vt, const unsigned char* __restrict__ mw,
    __bf16* __restrict__ O) {
  __shared__ __align__(16) __bf16 Pbuf[4][16 * 32];   // wave-private P tiles

  int blk = blockIdx.x;
  int qg = blk & 15, hh = (blk >> 4) & 7, win = blk >> 7;
  int tid = threadIdx.x, lane = tid & 31, wv = tid >> 5;
  int qbase = qg * 64 + wv * 16;
  int n = lane & 15, h = lane >> 4;

  const __bf16* Qp = Qb + ((size_t)(win * HH + hh)) * LTOK * 64;
  const __bf16* Kp = Kb + ((size_t)(win * HH + hh)) * LTOK * 64;
  const __bf16* Vp = Vt + ((size_t)(win * HH + hh)) * DH * LTOK;
  const unsigned char* mp = mw + (size_t)win * LTOK;

  v16bf A0 = ld_op(Qp, qbase, 64, 0, lane);    // q, d 0..31  (values part)
  v16bf A1 = ld_op(Qp, qbase, 64, 32, lane);   // q, d 32..63 (coords part)

  float m8[8], l8[8];
  v8f o0 = {}, o1 = {};
#pragma unroll
  for (int r = 0; r < 8; ++r) { m8[r] = -1e30f; l8[r] = 0.0f; }

  for (int kc = 0; kc < 32; ++kc) {            // 32 chunks x 32 keys = 1024
    int kb = kc * 32;
    v8f S0 = {}, S1 = {};
    S0 = wmma_bf16(A0, ld_op(Kp, kb,      64, 0,  lane), S0);
    S0 = wmma_bf16(A1, ld_op(Kp, kb,      64, 32, lane), S0);
    S1 = wmma_bf16(A0, ld_op(Kp, kb + 16, 64, 0,  lane), S1);
    S1 = wmma_bf16(A1, ld_op(Kp, kb + 16, 64, 32, lane), S1);

    bool mk0 = mp[kb + n] != 0;
    bool mk1 = mp[kb + 16 + n] != 0;

#pragma unroll
    for (int r = 0; r < 8; ++r) {
      float s0 = mk0 ? -1e30f : S0[r];
      float s1 = mk1 ? -1e30f : S1[r];
      float t = fmaxf(s0, s1);                 // row max across 32 keys
      t = fmaxf(t, __shfl_xor(t, 1, 32));
      t = fmaxf(t, __shfl_xor(t, 2, 32));
      t = fmaxf(t, __shfl_xor(t, 4, 32));
      t = fmaxf(t, __shfl_xor(t, 8, 32));
      float mn = fmaxf(m8[r], t);
      float alpha = __expf(m8[r] - mn);
      float p0 = __expf(s0 - mn);
      float p1 = __expf(s1 - mn);
      float rs = p0 + p1;
      rs += __shfl_xor(rs, 1, 32);
      rs += __shfl_xor(rs, 2, 32);
      rs += __shfl_xor(rs, 4, 32);
      rs += __shfl_xor(rs, 8, 32);
      l8[r] = l8[r] * alpha + rs;
      m8[r] = mn;
      o0[r] *= alpha;
      o1[r] *= alpha;
      int M = r + 8 * h;                       // C-layout -> A-layout via LDS
      Pbuf[wv][M * 32 + n]      = (__bf16)p0;
      Pbuf[wv][M * 32 + 16 + n] = (__bf16)p1;
    }
    asm volatile("s_wait_dscnt 0" ::: "memory");  // wave-private LDS RAW

    v16bf PA = ld_op(&Pbuf[wv][0], 0, 32, 0, lane);          // 16q x 32k
    o0 = wmma_bf16(PA, ld_op(Vp, 0,  LTOK, kb, lane), o0);   // d 0..15
    o1 = wmma_bf16(PA, ld_op(Vp, 16, LTOK, kb, lane), o1);   // d 16..31
  }

#pragma unroll
  for (int r = 0; r < 8; ++r) {
    float inv = (l8[r] > 0.0f) ? (1.0f / l8[r]) : 0.0f;
    int row = qbase + r + 8 * h;
    size_t base = ((size_t)win * LTOK + row) * INNERD + hh * DH;
    O[base + n]      = (__bf16)(o0[r] * inv);
    O[base + 16 + n] = (__bf16)(o1[r] * inv);
  }
}

// ---------- kernel 4: out projection + bias + roll/ungroup/unsort scatter ----------
__global__ __launch_bounds__(128) void k_out(
    const __bf16* __restrict__ O, const __bf16* __restrict__ woutT,
    const float* __restrict__ bout, const int* __restrict__ idx,
    float* __restrict__ out) {
  __shared__ __align__(16) __bf16 Ot[64 * 256];   // 32KB
  __shared__ __align__(16) __bf16 Wt[32 * 256];   // 16KB

  int blk = blockIdx.x;
  int win = blk / 16, mg = blk % 16;
  int b = win / WSZ, w = win % WSZ;
  int tid = threadIdx.x, lane = tid & 31, wv = tid >> 5;
  int rowBase = mg * 64;

  const __bf16* osrc = O + ((size_t)win * LTOK + rowBase) * INNERD;
  for (int e = tid; e < 64 * 256 / 8; e += blockDim.x)
    ((uint4*)Ot)[e] = ((const uint4*)osrc)[e];
  __syncthreads();

  for (int ng = 0; ng < 8; ++ng) {                // 256 cols / 32
    for (int e = tid; e < 32 * 256 / 8; e += blockDim.x)
      ((uint4*)Wt)[e] = ((const uint4*)(woutT + (size_t)ng * 32 * 256))[e];
    __syncthreads();
#pragma unroll
    for (int j = 0; j < 2; ++j) {
      int t8 = wv * 2 + j, mt = t8 >> 1, nt = t8 & 1;
      v8f acc = {};
#pragma unroll
      for (int kc = 0; kc < 8; ++kc)
        acc = wmma_bf16(ld_op(Ot, mt * 16, 256, kc * 32, lane),
                        ld_op(Wt, nt * 16, 256, kc * 32, lane), acc);
      int n = lane & 15, h = lane >> 4;
      int c = ng * 32 + nt * 16 + n;
      float bb = bout[c];
#pragma unroll
      for (int r = 0; r < 8; ++r) {
        int prow = rowBase + mt * 16 + r + 8 * h;     // token pos in window
        int jj = (prow + LTOK - 2) & (LTOK - 1);      // roll -ws/2 on (w t)
        int sp = w * WSZ + (jj >> 8);                 // sorted source slot
        int t2 = jj & 255;
        int s = idx[b * SS + sp];                     // unsort
        out[(((size_t)(b * SS + s)) * TT + t2) * DV + c] = acc[r] + bb;
      }
    }
    __syncthreads();
  }
}

extern "C" void kernel_launch(void* const* d_in, const int* in_sizes, int n_in,
                              void* d_out, int out_size, void* d_ws, size_t ws_size,
                              hipStream_t stream) {
  (void)in_sizes; (void)n_in; (void)out_size; (void)ws_size;
  const float* dt     = (const float*)d_in[0];
  const float* values = (const float*)d_in[1];
  const float* coords = (const float*)d_in[2];
  const unsigned char* mask = (const unsigned char*)d_in[3];
  const float* Wqkv = (const float*)d_in[4];
  const float* Wqkc = (const float*)d_in[5];
  const float* Wout = (const float*)d_in[6];
  const float* bout = (const float*)d_in[7];

  char* ws = (char*)d_ws;
  int* idx           = (int*)(ws + OFF_IDX);
  unsigned char* mw  = (unsigned char*)(ws + OFF_MW);
  __bf16* wqkvT = (__bf16*)(ws + OFF_WQKVT);
  __bf16* wqkcT = (__bf16*)(ws + OFF_WQKCT);
  __bf16* woutT = (__bf16*)(ws + OFF_WOUTT);
  __bf16* Qb    = (__bf16*)(ws + OFF_Q);
  __bf16* Kb    = (__bf16*)(ws + OFF_K);
  __bf16* Vtb   = (__bf16*)(ws + OFF_VT);
  __bf16* Ob    = (__bf16*)(ws + OFF_O);

  k_prep<<<1, 256, 0, stream>>>(dt, mask, idx, mw);

  int nw = 768 * 256 + 512 * 32 + 256 * 256;
  k_weights<<<(nw + 255) / 256, 256, 0, stream>>>(Wqkv, Wqkc, Wout,
                                                  wqkvT, wqkcT, woutT);

  k_proj<<<NWIN * 16, 128, 0, stream>>>(values, coords, idx, wqkvT, wqkcT,
                                        Qb, Kb, Vtb);

  k_attn<<<NWIN * HH * 16, 128, 0, stream>>>(Qb, Kb, Vtb, mw, Ob);

  k_out<<<NWIN * 16, 128, 0, stream>>>(Ob, woutT, bout, idx, (float*)d_out);
}
